// PatchCore_45956150067558
// MI455X (gfx1250) — matrix operand order repaired and checked by reading, not compiled, not run on hardware
//
#include <hip/hip_runtime.h>
#include <hip/hip_bf16.h>
#include <stdint.h>
#include <stddef.h>

// ---------------- problem constants ----------------
#define N_PATCH 676
#define NPAD    768            // 48 tiles of 16 (padded patch rows, zero-filled)
#define DIM     384
#define M_LIB   100000         // 6250 tiles of 16
#define BN      128            // patch rows per block (8 waves x 16)
#define BM      64             // bank rows per block (4 m-tiles)
#define KSTEP   32
#define KITERS  (DIM / KSTEP)  // 12
#define IMGSZ   224
#define FMAP    26
#define KS      33
#define GSIGMA  4.0f

#define USE_ASYNC_LDS 1

typedef __attribute__((ext_vector_type(16))) __bf16   v16bf;
typedef __attribute__((ext_vector_type(8)))  float    v8f;
typedef __attribute__((ext_vector_type(4)))  unsigned u32x4;
typedef __attribute__((ext_vector_type(2)))  unsigned u32x2;
typedef __attribute__((ext_vector_type(4)))  float    f32x4;

// ---------------- init ----------------
__global__ void __launch_bounds__(1024)
init_kernel(unsigned long long* minpack, unsigned long long* knnpack) {
    int t = threadIdx.x;
    if (t < NPAD) minpack[t] = ~0ull;
    if (t < 3)    knnpack[t] = ~0ull;
}

// ---------------- patch -> bf16 (padded) + x2 norms ----------------
__global__ void __launch_bounds__(256)
prep_patch_kernel(const float* __restrict__ patch, __bf16* __restrict__ pbf,
                  float* __restrict__ x2) {
    int wave = threadIdx.x >> 5, lane = threadIdx.x & 31;
    int r = blockIdx.x * 8 + wave;                 // 96 blocks * 8 waves = 768 rows
    float s = 0.f;
    for (int c = lane; c < DIM; c += 32) {
        float v = (r < N_PATCH) ? patch[(size_t)r * DIM + c] : 0.f;
        s += v * v;
        pbf[(size_t)r * DIM + c] = (__bf16)v;
    }
    for (int off = 16; off >= 1; off >>= 1) s += __shfl_xor(s, off);
    if (lane == 0) x2[r] = s;
}

// ---------------- bank y2 norms (also warms L2 with the bank) ----------------
__global__ void __launch_bounds__(256)
prep_y2_kernel(const float* __restrict__ lib, float* __restrict__ y2) {
    int wave = threadIdx.x >> 5, lane = threadIdx.x & 31;
    int r = blockIdx.x * 8 + wave;                 // 12500 blocks * 8 = 100000
    if (r >= M_LIB) return;
    float s = 0.f;
    for (int c = lane; c < DIM; c += 32) {
        float v = lib[(size_t)r * DIM + c];
        s += v * v;
    }
    for (int off = 16; off >= 1; off >>= 1) s += __shfl_xor(s, off);
    if (lane == 0) y2[r] = s;
}

// ---------------- main GEMM + fused min/argmin ----------------
__global__ void __launch_bounds__(256)
gemm_min_kernel(const __bf16* __restrict__ patchbf, const float* __restrict__ lib,
                const float* __restrict__ x2, const float* __restrict__ y2,
                unsigned long long* __restrict__ minpack) {
    __shared__ __bf16 Als[2][BN * KSTEP];   // 2 x 8KB
    __shared__ __bf16 Bls[2][BM * KSTEP];   // 2 x 4KB

    const int tid  = threadIdx.x;
    const int wave = tid >> 5;
    const int lane = tid & 31;
    const int half = lane >> 4;     // wave32: two 16-lane halves (WMMA layout)
    const int lrow = lane & 15;
    const int m0 = blockIdx.x * BM;
    const int n0 = blockIdx.y * BN;

    v8f acc[4] = {};

    auto load_chunk = [&](int kk, int buf) {
        const int k0 = kk * KSTEP;
        // A tile: 128 rows x 32 bf16 = 512 b128 pieces, 2 per thread, ASYNC to LDS
        #pragma unroll
        for (int r = 0; r < 2; ++r) {
            int idx = tid + r * 256;
            int row = idx >> 2;
            int qc  = idx & 3;
            const __bf16* g = patchbf + (size_t)(n0 + row) * DIM + k0 + qc * 8;
            __bf16* l = &Als[buf][row * KSTEP + qc * 8];
#if USE_ASYNC_LDS
            asm volatile("global_load_async_to_lds_b128 %0, %1, off"
                         :: "v"((unsigned)(uintptr_t)l),
                            "v"((unsigned long long)(uintptr_t)g)
                         : "memory");
#else
            *(u32x4*)l = *(const u32x4*)g;
#endif
        }
        // B tile: 64 rows x 32 f32 -> bf16 in LDS (on-the-fly conversion; bank
        // stays f32 in HBM/L2, no 77MB staging buffer)
        #pragma unroll
        for (int r = 0; r < 2; ++r) {
            int idx = tid + r * 256;
            int row = idx >> 3;
            int fc  = idx & 7;
            int gm  = m0 + row;
            f32x4 fv = {0.f, 0.f, 0.f, 0.f};
            if (gm < M_LIB) fv = *(const f32x4*)(lib + (size_t)gm * DIM + k0 + fc * 4);
            union { __bf16 h[4]; u32x2 u; } cv;
            #pragma unroll
            for (int i = 0; i < 4; ++i) cv.h[i] = (__bf16)fv[i];
            *(u32x2*)&Bls[buf][row * KSTEP + fc * 4] = cv.u;
        }
    };

    load_chunk(0, 0);
#if USE_ASYNC_LDS
    asm volatile("s_wait_asynccnt 0x0" ::: "memory");
#endif
    __syncthreads();

    for (int kk = 0; kk < KITERS; ++kk) {
        const int cur = kk & 1;
        if (kk + 1 < KITERS) load_chunk(kk + 1, cur ^ 1);

        const u32x4* As4 = (const u32x4*)&Als[cur][0];
        const u32x4* Bs4 = (const u32x4*)&Bls[cur][0];

        union Frag { v16bf v; u32x4 q[2]; } a, b;
        // A fragment 16x32: lane half 0 -> K 0..7 & 16..23, half 1 -> K 8..15 & 24..31
        {
            int arow = wave * 16 + lrow;
            a.q[0] = As4[arow * 4 + half];
            a.q[1] = As4[arow * 4 + half + 2];
        }
        #pragma unroll
        for (int t = 0; t < 4; ++t) {
            // B fragment 32x16: lane half 0 -> K 0..7 & 16..23, half 1 -> K 8..15 & 24..31
            int brow = t * 16 + lrow;
            b.q[0] = Bs4[brow * 4 + half];
            b.q[1] = Bs4[brow * 4 + half + 2];
            acc[t] = __builtin_amdgcn_wmma_f32_16x16x32_bf16(
                false, a.v, false, b.v, (short)0, acc[t], false, false);
        }

#if USE_ASYNC_LDS
        asm volatile("s_wait_asynccnt 0x0" ::: "memory");
#endif
        __syncthreads();
    }

    // epilogue: dist = sqrt(max(x2 + y2 - 2*dot, 0)); fused min+argmin via
    // monotone (float_bits<<32)|idx packing and u64 atomicMin.
    const int nb = n0 + wave * 16 + half * 8;      // C layout: vgpr j -> M=j (+8 upper half)
    #pragma unroll
    for (int j = 0; j < 8; ++j) {
        int n = nb + j;
        float xn = x2[n];
        unsigned long long best = ~0ull;
        #pragma unroll
        for (int t = 0; t < 4; ++t) {
            int m = m0 + t * 16 + lrow;
            if (m < M_LIB) {
                float d2 = xn + y2[m] - 2.0f * acc[t][j];
                float d  = sqrtf(fmaxf(d2, 0.0f));
                unsigned long long pk =
                    ((unsigned long long)__float_as_uint(d) << 32) | (unsigned)m;
                best = pk < best ? pk : best;
            }
        }
        #pragma unroll
        for (int off = 8; off >= 1; off >>= 1) {
            unsigned long long o = __shfl_xor(best, off);  // stays within 16-lane half
            best = o < best ? o : best;
        }
        if (lrow == 0) atomicMin(&minpack[n], best);
    }
}

// ---------------- argmax over per-row min -> s_idx, s_star, m_star ----------------
__global__ void __launch_bounds__(1024)
select_kernel(const unsigned long long* __restrict__ minpack, unsigned* __restrict__ sc) {
    __shared__ unsigned long long red[1024];
    int t = threadIdx.x;
    unsigned long long pk = 0ull;
    if (t < N_PATCH) {
        unsigned vb = (unsigned)(minpack[t] >> 32);
        pk = ((unsigned long long)vb << 32) | (unsigned)t;
    }
    red[t] = pk;
    __syncthreads();
    for (int s = 512; s > 0; s >>= 1) {
        if (t < s) { if (red[t + s] > red[t]) red[t] = red[t + s]; }
        __syncthreads();
    }
    if (t == 0) {
        unsigned long long b = red[0];
        unsigned sidx = (unsigned)(b & 0xffffffffu);
        sc[0] = sidx;                                   // s_idx
        sc[1] = (unsigned)(b >> 32);                    // s_star (float bits)
        sc[2] = (unsigned)(minpack[sidx] & 0xffffffffu);// m_star index in bank
    }
}

// ---------------- k-th nearest neighbor of m_star (exclusion pass) ----------------
__global__ void __launch_bounds__(256)
knn_kernel(const float* __restrict__ lib, const unsigned* __restrict__ sc,
           unsigned long long* __restrict__ knnpack, int k) {
    int wave = threadIdx.x >> 5, lane = threadIdx.x & 31;
    int r = blockIdx.x * 8 + wave;
    if (r >= M_LIB) return;
    unsigned star = sc[2];
    const float* a = lib + (size_t)star * DIM;
    const float* b = lib + (size_t)r * DIM;
    float s = 0.f;
    for (int c = lane; c < DIM; c += 32) { float d = a[c] - b[c]; s += d * d; }
    for (int off = 16; off >= 1; off >>= 1) s += __shfl_xor(s, off);
    if (lane == 0) {
        bool excl = false;
        for (int j = 0; j < k; ++j)
            if ((unsigned)(knnpack[j] & 0xffffffffu) == (unsigned)r) excl = true;
        if (!excl) {
            float d = sqrtf(s);
            unsigned long long pk =
                ((unsigned long long)__float_as_uint(d) << 32) | (unsigned)r;
            atomicMin(&knnpack[k], pk);
        }
    }
}

// ---------------- scalar score: softmax reweighting ----------------
__global__ void __launch_bounds__(64)
fin_kernel(const float* __restrict__ patch, const float* __restrict__ lib,
           const unsigned* __restrict__ sc, const unsigned long long* __restrict__ knnpack,
           float* __restrict__ out) {
    __shared__ float dd[2];
    int wave = threadIdx.x >> 5, lane = threadIdx.x & 31;
    unsigned sidx = sc[0];
    unsigned nn = (unsigned)(knnpack[1 + wave] & 0xffffffffu);  // nn_idx[1:], dist to m_test
    const float* a = patch + (size_t)sidx * DIM;
    const float* b = lib + (size_t)nn * DIM;
    float s = 0.f;
    for (int c = lane; c < DIM; c += 32) { float d = a[c] - b[c]; s += d * d; }
    for (int off = 16; off >= 1; off >>= 1) s += __shfl_xor(s, off);
    if (lane == 0) dd[wave] = sqrtf(s);
    __syncthreads();
    if (threadIdx.x == 0) {
        float sstar = __uint_as_float(sc[1]);
        float sq = sqrtf((float)DIM);
        float den = expf(dd[0] / sq) + expf(dd[1] / sq);
        float w = 1.0f - expf(sstar / sq) / den;
        out[0] = w * sstar;
    }
}

// ---------------- bilinear upsample 26x26 -> 224x224 (align_corners=False) ----------------
__global__ void __launch_bounds__(256)
upsample_kernel(const unsigned long long* __restrict__ minpack, float* __restrict__ up) {
    int g = blockIdx.x * 256 + threadIdx.x;        // 196*256 == 50176 exactly
    int oy = g / IMGSZ, ox = g % IMGSZ;
    const float scale = (float)FMAP / (float)IMGSZ;
    float fy = (oy + 0.5f) * scale - 0.5f;
    float fx = (ox + 0.5f) * scale - 0.5f;
    int y0 = (int)floorf(fy), x0 = (int)floorf(fx);
    float wy = fy - (float)y0, wx = fx - (float)x0;
    int y0c = min(max(y0, 0), FMAP - 1), y1c = min(max(y0 + 1, 0), FMAP - 1);
    int x0c = min(max(x0, 0), FMAP - 1), x1c = min(max(x0 + 1, 0), FMAP - 1);
    auto val = [&](int yy, int xx) {
        return __uint_as_float((unsigned)(minpack[yy * FMAP + xx] >> 32));
    };
    float v = (1.f - wy) * ((1.f - wx) * val(y0c, x0c) + wx * val(y0c, x1c)) +
              wy * ((1.f - wx) * val(y1c, x0c) + wx * val(y1c, x1c));
    up[g] = v;
}

// ---------------- separable 33-tap Gaussian, reflect padding ----------------
__global__ void __launch_bounds__(256)
blur_kernel(const float* __restrict__ src, float* __restrict__ dst, int horiz) {
    __shared__ float wl[KS];
    int t = threadIdx.x;
    if (t < KS) {
        float x = (float)t - (KS - 1) * 0.5f;
        wl[t] = expf(-0.5f * (x / GSIGMA) * (x / GSIGMA));
    }
    __syncthreads();
    float wsum = 0.f;
    for (int i = 0; i < KS; ++i) wsum += wl[i];
    int g = blockIdx.x * 256 + t;
    int oy = g / IMGSZ, ox = g % IMGSZ;
    float acc = 0.f;
    for (int i = 0; i < KS; ++i) {
        int p = (horiz ? ox : oy) - (KS - 1) / 2 + i;
        if (p < 0) p = -p;                               // jnp.pad 'reflect'
        if (p > IMGSZ - 1) p = 2 * (IMGSZ - 1) - p;
        acc += wl[i] * (horiz ? src[oy * IMGSZ + p] : src[p * IMGSZ + ox]);
    }
    dst[g] = acc / wsum;
}

// ---------------- host launcher ----------------
extern "C" void kernel_launch(void* const* d_in, const int* in_sizes, int n_in,
                              void* d_out, int out_size, void* d_ws, size_t ws_size,
                              hipStream_t stream) {
    (void)in_sizes; (void)n_in; (void)out_size; (void)ws_size;
    const float* patch = (const float*)d_in[0];   // [676, 384]
    const float* lib   = (const float*)d_in[1];   // [100000, 384]
    float* out = (float*)d_out;                   // [1 + 224*224]

    char* base = (char*)d_ws;
    size_t off = 0;
    auto take = [&](size_t bytes) -> void* {
        void* p = base + off;
        off = (off + bytes + 255) & ~(size_t)255;
        return p;
    };
    __bf16* pbf               = (__bf16*)take((size_t)NPAD * DIM * sizeof(__bf16));
    float* x2                 = (float*)take((size_t)NPAD * sizeof(float));
    float* y2                 = (float*)take((size_t)M_LIB * sizeof(float));
    unsigned long long* minpk = (unsigned long long*)take((size_t)NPAD * 8);
    unsigned long long* knnpk = (unsigned long long*)take(3 * 8);
    unsigned* sc              = (unsigned*)take(8 * sizeof(unsigned));
    float* up                 = (float*)take((size_t)IMGSZ * IMGSZ * sizeof(float));
    float* tmp                = (float*)take((size_t)IMGSZ * IMGSZ * sizeof(float));

    init_kernel<<<1, 1024, 0, stream>>>(minpk, knnpk);
    prep_patch_kernel<<<NPAD / 8, 256, 0, stream>>>(patch, pbf, x2);
    prep_y2_kernel<<<M_LIB / 8, 256, 0, stream>>>(lib, y2);

    dim3 ggrid((M_LIB + BM - 1) / BM, NPAD / BN, 1);   // 1563 x 6
    gemm_min_kernel<<<ggrid, 256, 0, stream>>>(pbf, lib, x2, y2, minpk);

    select_kernel<<<1, 1024, 0, stream>>>(minpk, sc);
    for (int k = 0; k < 3; ++k)
        knn_kernel<<<M_LIB / 8, 256, 0, stream>>>(lib, sc, knnpk, k);
    fin_kernel<<<1, 64, 0, stream>>>(patch, lib, sc, knnpk, out);

    upsample_kernel<<<(IMGSZ * IMGSZ) / 256, 256, 0, stream>>>(minpk, up);
    blur_kernel<<<(IMGSZ * IMGSZ) / 256, 256, 0, stream>>>(up, tmp, 1);
    blur_kernel<<<(IMGSZ * IMGSZ) / 256, 256, 0, stream>>>(tmp, out + 1, 0);
}
